// Attention_53558242181469
// MI455X (gfx1250) — matrix operand order repaired and checked by reading, not compiled, not run on hardware
//
#include <hip/hip_runtime.h>

// ---------------------------------------------------------------------------
// CDNA5 (gfx1250) fused attention pipeline, bf16 WMMA everywhere.
// LDS-tiled GEMMs (64x64 per WG) with async global->LDS panel staging,
// fused LDS-resident attention core.
// ---------------------------------------------------------------------------

typedef __attribute__((ext_vector_type(16))) __bf16 v16bf;
typedef __attribute__((ext_vector_type(8)))  float  v8f;
typedef __attribute__((ext_vector_type(4)))  float  f32x4;

__device__ __forceinline__ v8f wmma_bf16(v16bf a, v16bf b, v8f c) {
  // D = A x B + C, 16x16x32 bf16 -> f32
  return __builtin_amdgcn_wmma_f32_16x16x32_bf16(
      /*neg_a=*/false, a, /*neg_b=*/false, b,
      /*c_mod=*/(short)0, c, /*reuse_a=*/false, /*reuse_b=*/false);
}

// --- async global->LDS copy (CDNA5 GLOBAL_LOAD_ASYNC_TO_LDS_B128) ----------
// Per-lane: copies 16 bytes from global address vaddr to LDS address vdst, tracked by ASYNCcnt.
// LDS aperture keeps the LDS byte offset in addr[31:0] (ISA 10.2), so the
// truncated generic pointer is a valid VDST LDS address.
__device__ __forceinline__ void async_copy16(void* lds_dst, const void* gsrc) {
  unsigned lds_off = (unsigned)(unsigned long long)lds_dst;
  asm volatile("global_load_async_to_lds_b128 %0, %1, off"
               :: "v"(lds_off), "v"(gsrc)
               : "memory");
}
__device__ __forceinline__ void async_wait0() {
  asm volatile("s_wait_asynccnt 0" ::: "memory");
}

// --- fragment loaders (contiguous k order: k = 16*(lane>>4) + j) ---------------------------------
// A (16x32, 16-bit): row = lane&15; lanes<16 hold K 0..15, lanes>=16 K 16..31.
__device__ __forceinline__ v16bf load_a_f32(const float* base, int ld, int k0, int lane) {
  v16bf a;
  const float* row = base + (size_t)(lane & 15) * ld + k0 + ((lane & 16) ? 16 : 0);
#pragma unroll
  for (int j = 0; j < 16; ++j) a[j] = (__bf16)row[j];
  return a;
}
__device__ __forceinline__ v16bf load_a_bf16(const __bf16* base, int ld, int k0, int lane) {
  v16bf a;
  const __bf16* row = base + (size_t)(lane & 15) * ld + k0 + ((lane & 16) ? 16 : 0);
#pragma unroll
  for (int j = 0; j < 16; ++j) a[j] = row[j];
  return a;
}
// B (32x16, 16-bit): n = lane&15; lanes<16 hold K 0..15, lanes>=16 K 16..31.
// source layout B[k][n] = src[n*ldn + k]  (i.e. multiplying by W^T)
__device__ __forceinline__ v16bf load_b_nk_f32(const float* base, int ldn, int k0, int lane) {
  v16bf b;
  const float* col = base + (size_t)(lane & 15) * ldn + k0 + ((lane & 16) ? 16 : 0);
#pragma unroll
  for (int j = 0; j < 16; ++j) b[j] = (__bf16)col[j];
  return b;
}
__device__ __forceinline__ v16bf load_b_nk_bf16(const __bf16* base, int ldn, int k0, int lane) {
  v16bf b;
  const __bf16* col = base + (size_t)(lane & 15) * ldn + k0 + ((lane & 16) ? 16 : 0);
#pragma unroll
  for (int j = 0; j < 16; ++j) b[j] = col[j];
  return b;
}
// source layout B[k][n] = src[k*ldk + n]
__device__ __forceinline__ v16bf load_b_kn_f32(const float* base, int ldk, int k0, int lane) {
  v16bf b;
  const float* p = base + (size_t)(k0 + ((lane & 16) ? 16 : 0)) * ldk + (lane & 15);
#pragma unroll
  for (int j = 0; j < 16; ++j) b[j] = (__bf16)p[(size_t)j * ldk];
  return b;
}
// B fragment of the skew matrix S = wedge_A - wedge_A^T + diag(wedge_bias[h])
__device__ __forceinline__ v16bf load_b_S(const float* wA, const float* wb_h,
                                          int n0, int k0, int lane) {
  v16bf b;
  int e = n0 + (lane & 15);
  int kb = k0 + ((lane & 16) ? 16 : 0);
#pragma unroll
  for (int j = 0; j < 16; ++j) {
    int d = kb + j;
    float s = wA[d * 64 + e] - wA[e * 64 + d] + (d == e ? wb_h[d] : 0.f);
    b[j] = (__bf16)s;
  }
  return b;
}

// ---------------------------------------------------------------------------
// 1) LDS-tiled GEMM:  Y[M,N] = X[M,K] @ W[N,K]^T + bias[N]   (f32 in/out)
//    64x64 output tile per workgroup; f32 panels staged via async global->LDS,
//    converted to bf16 inside the fragment loaders.
//    wave w: fixed n-subtile (w&3), two m-subtiles (w>>2, (w>>2)+2).
// ---------------------------------------------------------------------------
__global__ void gemm_xwt_tiled(const float* __restrict__ X, const float* __restrict__ W,
                               const float* __restrict__ bias, float* __restrict__ Y,
                               int M, int N, int K) {
  __shared__ float Asf[64 * 32];  // [m][k]
  __shared__ float Bsf[64 * 32];  // [n][k]
  int lane = threadIdx.x & 31;
  int wv = threadIdx.x >> 5;          // 0..7
  int tilesN = N >> 6;
  int m0 = ((int)blockIdx.x / tilesN) << 6;
  int n0 = ((int)blockIdx.x % tilesN) << 6;

  int ncol = wv & 3;
  int mrow0 = wv >> 2;                // 0..1 -> sub-rows mrow0, mrow0+2
  v8f acc0 = {}, acc1 = {};

  int lrow = threadIdx.x >> 2;        // 0..63
  int lcol = (threadIdx.x & 3) << 3;  // 0,8,16,24

  for (int k0 = 0; k0 < K; k0 += 32) {
    const float* ga = X + (size_t)(m0 + lrow) * K + k0 + lcol;
    const float* gb = W + (size_t)(n0 + lrow) * K + k0 + lcol;
    if (k0 + 32 < K) {                // prefetch next k-panel -> global_prefetch_b8
      __builtin_prefetch(ga + 32, 0, 1);
      __builtin_prefetch(gb + 32, 0, 1);
    }
    float* pa = Asf + lrow * 32 + lcol;
    float* pb = Bsf + lrow * 32 + lcol;
    async_copy16(pa, ga);
    async_copy16(pa + 4, ga + 4);
    async_copy16(pb, gb);
    async_copy16(pb + 4, gb + 4);
    async_wait0();
    __syncthreads();
    v16bf bfr  = load_b_nk_f32(Bsf + (size_t)(ncol * 16) * 32, 32, 0, lane);
    v16bf afr0 = load_a_f32(Asf + (size_t)(mrow0 * 16) * 32, 32, 0, lane);
    v16bf afr1 = load_a_f32(Asf + (size_t)((mrow0 + 2) * 16) * 32, 32, 0, lane);
    acc0 = wmma_bf16(afr0, bfr, acc0);
    acc1 = wmma_bf16(afr1, bfr, acc1);
    __syncthreads();
  }
  int n = n0 + ncol * 16 + (lane & 15);
  float bv = bias[n];
#pragma unroll
  for (int r = 0; r < 8; ++r) {
    int mloc = r + ((lane & 16) ? 8 : 0);
    Y[(size_t)(m0 + mrow0 * 16 + mloc) * N + n] = acc0[r] + bv;
    Y[(size_t)(m0 + (mrow0 + 2) * 16 + mloc) * N + n] = acc1[r] + bv;
  }
}

// ---------------------------------------------------------------------------
// 2) RMS-norm of 64-element rows (q heads), one wave per row.
// ---------------------------------------------------------------------------
__global__ void rmsnorm64(float* __restrict__ q, int nseg) {
  int seg = blockIdx.x * (blockDim.x >> 5) + (threadIdx.x >> 5);
  if (seg >= nseg) return;
  int lane = threadIdx.x & 31;
  float* p = q + (size_t)seg * 64;
  float x0 = p[lane], x1 = p[lane + 32];
  float ss = x0 * x0 + x1 * x1;
#pragma unroll
  for (int o = 16; o > 0; o >>= 1) ss += __shfl_xor(ss, o, 32);
  float r = rsqrtf(ss * (1.f / 64.f) + 1.192092896e-07f);
  p[lane] = x0 * r;
  p[lane + 32] = x1 * r;
}

// ---------------------------------------------------------------------------
// 3) x <- RoPE(x + x @ S_h)  per (b,h,16-token tile).  WMMA: C preloaded
//    with x so the "+x" rides in the accumulator.  Output bf16 [B,48,T,64].
// ---------------------------------------------------------------------------
__global__ void srope_kernel(const float* __restrict__ in,
                             const float* __restrict__ wedgeA,
                             const float* __restrict__ wedgeB,
                             __bf16* __restrict__ out, int headsIn) {
  __shared__ float tile[16 * 64];
  int lane = threadIdx.x & 31;
  int wv = threadIdx.x >> 5;  // 0..3, one 16x16 N-tile each
  int idx = blockIdx.x;
  int tb = idx & 63;
  int h = (idx >> 6) % 48;
  int b = idx / (64 * 48);
  int t0 = tb << 4;
  int hin = h % headsIn;
  int ld = headsIn * 64;
  const float* base = in + (size_t)(b * 1024 + t0) * ld + hin * 64;
  int n0 = wv << 4;

  v8f acc;
#pragma unroll
  for (int r = 0; r < 8; ++r)
    acc[r] = base[(size_t)(r + ((lane & 16) ? 8 : 0)) * ld + n0 + (lane & 15)];
  v16bf a0 = load_a_f32(base, ld, 0, lane);
  v16bf a1 = load_a_f32(base, ld, 32, lane);
  v16bf b0 = load_b_S(wedgeA, wedgeB + h * 64, n0, 0, lane);
  v16bf b1 = load_b_S(wedgeA, wedgeB + h * 64, n0, 32, lane);
  acc = wmma_bf16(a0, b0, acc);
  acc = wmma_bf16(a1, b1, acc);
#pragma unroll
  for (int r = 0; r < 8; ++r) {
    int m = r + ((lane & 16) ? 8 : 0);
    tile[m * 64 + n0 + (lane & 15)] = acc[r];
  }
  __syncthreads();

  __bf16* obase = out + ((size_t)(b * 48 + h) * 1024 + t0) * 64;
  for (int i = threadIdx.x; i < 16 * 32; i += blockDim.x) {
    int row = i >> 5, e = i & 31;
    float x1 = tile[row * 64 + 2 * e];
    float x2 = tile[row * 64 + 2 * e + 1];
    float ang = (float)(t0 + row) * __expf(-(2.f * (float)e / 64.f) * 9.2103403720f);
    float c = __cosf(ang), s = __sinf(ang);
    obase[(size_t)row * 64 + e] = (__bf16)(x1 * c - x2 * s);
    obase[(size_t)row * 64 + e + 32] = (__bf16)(x1 * s + x2 * c);
  }
}

// ---------------------------------------------------------------------------
// 4) Fused attention: one workgroup per (b,h,16-token block).
//    QK^T (WMMA) -> LDS scores -> softmax+sink -> top-12 -> marker -> MLP
//    (WMMA fc/proj) -> context (bf16).  Scores never hit HBM.
// ---------------------------------------------------------------------------
__global__ void attn_kernel(const __bf16* __restrict__ qf, const __bf16* __restrict__ kf,
                            const float* __restrict__ kp, const float* __restrict__ sinks,
                            const float* __restrict__ v_nulls,
                            const float* __restrict__ fc_w, const float* __restrict__ fc_b,
                            const float* __restrict__ proj_w, const float* __restrict__ proj_b,
                            __bf16* __restrict__ ctx) {
  extern __shared__ char smem[];
  float* scores = (float*)smem;         // 16 x 1024
  float* h1 = scores + 16 * 1024;       // 16 x 256
  float* marker = h1 + 16 * 256;        // 16 x 64
  float* psink = marker + 16 * 64;      // 16
  __bf16* hb = (__bf16*)(psink + 16);   // 16 x 256 bf16

  int lane = threadIdx.x & 31;
  int wv = threadIdx.x >> 5;  // 0..7
  int idx = blockIdx.x;
  int tb = idx & 63;
  int h = (idx >> 6) % 48;
  int b = idx / (64 * 48);
  int t0 = tb << 4;
  int nh = h % 12;

  // ---- phase 1: scores = (q . k) * DH^-0.5, causal mask, into LDS --------
  const __bf16* qbase = qf + ((size_t)(b * 48 + h) * 1024 + t0) * 64;
  v16bf a0 = load_a_bf16(qbase, 64, 0, lane);
  v16bf a1 = load_a_bf16(qbase, 64, 32, lane);
  for (int st = wv; st < 64; st += 8) {
    int s0 = st << 4;
    const __bf16* kbase = kf + ((size_t)(b * 48 + h) * 1024 + s0) * 64;
    v16bf b0 = load_b_nk_bf16(kbase, 64, 0, lane);
    v16bf b1 = load_b_nk_bf16(kbase, 64, 32, lane);
    v8f acc = {};
    acc = wmma_bf16(a0, b0, acc);
    acc = wmma_bf16(a1, b1, acc);
    int s = s0 + (lane & 15);
#pragma unroll
    for (int r = 0; r < 8; ++r) {
      int m = r + ((lane & 16) ? 8 : 0);
      float v = acc[r] * 0.125f;
      if (s > t0 + m) v = -3.0e38f;
      scores[m * 1024 + s] = v;
    }
  }
  __syncthreads();

  // ---- phase 2: softmax(+sink), top-12, marker ---------------------------
  float sinkv = sinks[h];
  const float* kvan = kp + nh * 64;  // k_vanilla[b,h,s,d] = kp[(b*1024+s)*768 + nh*64 + d]
  for (int rr = 0; rr < 2; ++rr) {
    int row = wv * 2 + rr;
    float* srow = scores + row * 1024;
    float mx = -3.0e38f;
    for (int s = lane; s < 1024; s += 32) mx = fmaxf(mx, srow[s]);
#pragma unroll
    for (int o = 16; o > 0; o >>= 1) mx = fmaxf(mx, __shfl_xor(mx, o, 32));
    mx = fmaxf(mx, sinkv);
    float sum = 0.f;
    for (int s = lane; s < 1024; s += 32) sum += __expf(srow[s] - mx);
#pragma unroll
    for (int o = 16; o > 0; o >>= 1) sum += __shfl_xor(sum, o, 32);
    sum += __expf(sinkv - mx);

    float acc0 = 0.f, acc1 = 0.f;
    for (int kk = 0; kk < 12; ++kk) {
      float bv = -3.0e38f;
      int bi = 0;
      for (int s = lane; s < 1024; s += 32) {
        float v = srow[s];
        if (v > bv) { bv = v; bi = s; }
      }
#pragma unroll
      for (int o = 16; o > 0; o >>= 1) {
        float ov = __shfl_xor(bv, o, 32);
        int oi = __shfl_xor(bi, o, 32);
        if (ov > bv || (ov == bv && oi < bi)) { bv = ov; bi = oi; }
      }
      float w = __expf(bv - mx) / sum;  // masked picks -> weight 0
      const float* krow = kvan + (size_t)(b * 1024 + bi) * 768;
      acc0 += w * krow[lane];
      acc1 += w * krow[lane + 32];
      if (lane == 0) srow[bi] = -3.0e38f;
      __syncthreads();  // uniform across all 8 waves
    }
    {
      const float* krow = kvan + (size_t)(b * 1024 + t0 + row) * 768;
      marker[row * 64 + lane] = (acc0 + krow[lane]) * (1.f / 13.f);
      marker[row * 64 + lane + 32] = (acc1 + krow[lane + 32]) * (1.f / 13.f);
      if (lane == 0) psink[row] = __expf(sinkv - mx) / sum;
    }
  }
  __syncthreads();

  // ---- phase 3: MLP (fc 64->256, act, rmsnorm, silu, proj 256->64) -------
  v16bf ma0 = load_a_f32(marker, 64, 0, lane);
  v16bf ma1 = load_a_f32(marker, 64, 32, lane);
  for (int nt = wv; nt < 16; nt += 8) {
    int n0 = nt << 4;
    v16bf b0 = load_b_nk_f32(fc_w + (size_t)n0 * 64, 64, 0, lane);
    v16bf b1 = load_b_nk_f32(fc_w + (size_t)n0 * 64, 64, 32, lane);
    v8f acc = {};
    acc = wmma_bf16(ma0, b0, acc);
    acc = wmma_bf16(ma1, b1, acc);
#pragma unroll
    for (int r = 0; r < 8; ++r) {
      int m = r + ((lane & 16) ? 8 : 0);
      int n = n0 + (lane & 15);
      float v = acc[r] + fc_b[n];
      float v2 = v * v;
      h1[m * 256 + n] = v2 + 0.75f * v2 * v;
    }
  }
  __syncthreads();
  for (int rr = 0; rr < 2; ++rr) {
    int row = wv * 2 + rr;
    float ss = 0.f;
    for (int j = lane; j < 256; j += 32) { float v = h1[row * 256 + j]; ss += v * v; }
#pragma unroll
    for (int o = 16; o > 0; o >>= 1) ss += __shfl_xor(ss, o, 32);
    float rn = rsqrtf(ss * (1.f / 256.f) + 1.192092896e-07f);
    for (int j = lane; j < 256; j += 32) {
      float v = h1[row * 256 + j] * rn;
      float sg = 1.f / (1.f + __expf(-1.8137993642f * v));  // pi/sqrt(3)
      hb[row * 256 + j] = (__bf16)(v * sg);
    }
  }
  __syncthreads();
  if (wv < 4) {
    int n0 = wv << 4;
    v8f acc = {};
    for (int kk = 0; kk < 256; kk += 32) {
      v16bf a = load_a_bf16(hb, 256, kk, lane);
      v16bf bb = load_b_nk_f32(proj_w + (size_t)n0 * 256, 256, kk, lane);
      acc = wmma_bf16(a, bb, acc);
    }
#pragma unroll
    for (int r = 0; r < 8; ++r) {
      int m = r + ((lane & 16) ? 8 : 0);
      int n = n0 + (lane & 15);
      float v = acc[r] + proj_b[n] + psink[m] * v_nulls[h * 64 + n];
      ctx[((size_t)(b * 4 + h / 12) * 1024 + t0 + m) * 768 + nh * 64 + n] = (__bf16)v;
    }
  }
}

// ---------------------------------------------------------------------------
// 5) y[b,t,d] = mean_br( ctx[b,br,t,:] @ WO[br,:,d] + WO_b[br,d] )
//    LDS-tiled 64x64 per workgroup, br accumulated in registers.
//    A panel (bf16) and B panel (f32) staged via async global->LDS.
// ---------------------------------------------------------------------------
__global__ void final_gemm_tiled(const __bf16* __restrict__ ctx, const float* __restrict__ WO,
                                 const float* __restrict__ WOb, float* __restrict__ out) {
  __shared__ __bf16 As[64 * 32];  // [m][k] bf16
  __shared__ float Bsf[32 * 64];  // [k][n] f32
  int lane = threadIdx.x & 31;
  int wv = threadIdx.x >> 5;
  int tile = blockIdx.x;          // per b: (1024/64)*(768/64) = 16*12 = 192
  int b = tile / 192;
  int rem = tile % 192;
  int m0 = (rem / 12) << 6;
  int n0 = (rem % 12) << 6;
  int ncol = wv & 3;
  int mrow0 = wv >> 2;
  v8f acc0 = {}, acc1 = {};

  int lrowA = threadIdx.x >> 2;        // 0..63
  int lcolA = (threadIdx.x & 3) << 3;  // 0,8,16,24
  int krowB = threadIdx.x >> 3;        // 0..31
  int ncolB = (threadIdx.x & 7) << 3;  // 0..56

  for (int br = 0; br < 4; ++br) {
    const __bf16* Abase = ctx + ((size_t)(b * 4 + br) * 1024 + m0) * 768;
    const float* Bbase = WO + (size_t)br * 768 * 768 + n0;
    for (int k0 = 0; k0 < 768; k0 += 32) {
      // stage A panel (raw bf16 bytes, one async b128 per thread)
      async_copy16(As + lrowA * 32 + lcolA, Abase + (size_t)lrowA * 768 + k0 + lcolA);
      // stage B panel (f32, two async b128 per thread)
      const float* gb = Bbase + (size_t)(k0 + krowB) * 768 + ncolB;
      if (k0 + 32 < 768) __builtin_prefetch(gb + 32 * 768, 0, 1);
      float* pb = Bsf + krowB * 64 + ncolB;
      async_copy16(pb, gb);
      async_copy16(pb + 4, gb + 4);
      async_wait0();
      __syncthreads();
      v16bf bfr = load_b_kn_f32(Bsf + ncol * 16, 64, 0, lane);
      v16bf a0f = load_a_bf16(As + (size_t)(mrow0 * 16) * 32, 32, 0, lane);
      v16bf a1f = load_a_bf16(As + (size_t)((mrow0 + 2) * 16) * 32, 32, 0, lane);
      acc0 = wmma_bf16(a0f, bfr, acc0);
      acc1 = wmma_bf16(a1f, bfr, acc1);
      __syncthreads();
    }
  }
  int n = n0 + ncol * 16 + (lane & 15);
  float bias = 0.25f * (WOb[n] + WOb[768 + n] + WOb[1536 + n] + WOb[2304 + n]);
#pragma unroll
  for (int r = 0; r < 8; ++r) {
    int mloc = r + ((lane & 16) ? 8 : 0);
    out[((size_t)b * 1024 + m0 + mrow0 * 16 + mloc) * 768 + n] = 0.25f * acc0[r] + bias;
    out[((size_t)b * 1024 + m0 + (mrow0 + 2) * 16 + mloc) * 768 + n] = 0.25f * acc1[r] + bias;
  }
}

// ---------------------------------------------------------------------------
extern "C" void kernel_launch(void* const* d_in, const int* in_sizes, int n_in,
                              void* d_out, int out_size, void* d_ws, size_t ws_size,
                              hipStream_t stream) {
  const float* A      = (const float*)d_in[0];
  const float* X      = (const float*)d_in[1];
  const float* Wq_w   = (const float*)d_in[2];
  const float* Wq_b   = (const float*)d_in[3];
  const float* Wk_w   = (const float*)d_in[4];
  const float* Wk_b   = (const float*)d_in[5];
  const float* wedgeA = (const float*)d_in[6];
  const float* wedgeB = (const float*)d_in[7];
  const float* sinks  = (const float*)d_in[8];
  const float* vnull  = (const float*)d_in[9];
  const float* fc_w   = (const float*)d_in[10];
  const float* fc_b   = (const float*)d_in[11];
  const float* proj_w = (const float*)d_in[12];
  const float* proj_b = (const float*)d_in[13];
  const float* WO     = (const float*)d_in[14];
  const float* WOb    = (const float*)d_in[15];
  float* out = (float*)d_out;

  // workspace carve (~69 MB)
  char* w = (char*)d_ws;
  float* qp = (float*)w;  w += (size_t)2048 * 3072 * sizeof(float);
  float* kp = (float*)w;  w += (size_t)2048 * 768 * sizeof(float);
  __bf16* qf = (__bf16*)w; w += (size_t)2 * 48 * 1024 * 64 * sizeof(__bf16);
  __bf16* kf = (__bf16*)w; w += (size_t)2 * 48 * 1024 * 64 * sizeof(__bf16);
  __bf16* ctx = (__bf16*)w;

  // 1) projections (64x64 LDS-tiled WMMA, async panel staging)
  gemm_xwt_tiled<<<32 * 48, 256, 0, stream>>>(A, Wq_w, Wq_b, qp, 2048, 3072, 768);
  gemm_xwt_tiled<<<32 * 12, 256, 0, stream>>>(X, Wk_w, Wk_b, kp, 2048, 768, 768);
  // 2) rmsnorm(q) over head dim
  rmsnorm64<<<(98304 + 7) / 8, 256, 0, stream>>>(qp, 98304);
  // 3) skew + rope -> bf16 q/k
  srope_kernel<<<2 * 48 * 64, 128, 0, stream>>>(qp, wedgeA, wedgeB, qf, 48);
  srope_kernel<<<2 * 48 * 64, 128, 0, stream>>>(kp, wedgeA, wedgeB, kf, 12);
  // 4) fused attention + top-k + MLP (scores live in LDS only)
  size_t smem = (size_t)16 * 1024 * 4 + 16 * 256 * 4 + 16 * 64 * 4 + 16 * 4 + 16 * 256 * 2;
  attn_kernel<<<2 * 48 * 64, 256, smem, stream>>>(qf, kf, kp, sinks, vnull,
                                                  fc_w, fc_b, proj_w, proj_b, ctx);
  // 5) output projection + branch mean (LDS-tiled, async staging)
  final_gemm_tiled<<<2 * 192, 256, 0, stream>>>(ctx, WO, WOb, out);
}
